// CRFLoss_vb_34162169872845
// MI455X (gfx1250) — matrix operand order repaired and checked by reading, not compiled, not run on hardware
//
#include <hip/hip_runtime.h>
#include <stdint.h>

#define SEQ_LEN   256
#define BAT       128
#define TAGS      64
#define START_TAG 62
#define END_TAG   63
#define CH        4                      // timesteps per TDM chunk
#define NCHUNK    (SEQ_LEN / CH)
#define ROW_DW    65                     // padded row stride (dwords) -> conflict-free LDS banks
#define TILE_DW   (TAGS * ROW_DW)        // 4160 dwords per timestep tile
#define CHUNK_DW  (CH * TILE_DW)

typedef _Float16 half_t;
typedef __attribute__((ext_vector_type(16))) _Float16     v16h;
typedef __attribute__((ext_vector_type(8)))  float        v8f;
typedef __attribute__((ext_vector_type(4)))  unsigned int u32x4;
typedef __attribute__((ext_vector_type(8)))  int          i32x8;
typedef __attribute__((ext_vector_type(4)))  int          i32x4;

#if __has_builtin(__builtin_amdgcn_tensor_load_to_lds)
#define HAVE_TDM 1
#else
#define HAVE_TDM 0
#endif

#if HAVE_TDM
// Build a 2D D# : tile = CH rows of 4096 f32 (one row per timestep, stride BAT*4096
// elements between timesteps), LDS destination padded 1 dword per 64 dwords so each
// 64-wide tag row lands at stride ROW_DW=65.
__device__ __forceinline__ void tdm_issue(const float* scores, int b, int t0, void* lds_dst) {
  const unsigned long long ga =
      (unsigned long long)(uintptr_t)scores +
      ((unsigned long long)t0 * BAT + (unsigned long long)b) * (TAGS * TAGS * sizeof(float));
  u32x4 g0;
  g0[0] = 1u;                                              // count=1, is_restore=0, gather off
  g0[1] = (unsigned)(uintptr_t)lds_dst;                    // LDS byte offset
  g0[2] = (unsigned)(ga & 0xFFFFFFFFull);                  // global_addr[31:0]
  g0[3] = (unsigned)((ga >> 32) & 0x1FFFFFFull) | (2u << 30); // global_addr[56:32] | type=2
  i32x8 g1;
  g1[0] = (int)((2u << 16) | (1u << 20) | (5u << 22));     // data_size=4B, pad_en, interval=64dw, amount=1dw
  g1[1] = (int)(4096u << 16);                              // tensor_dim0 = 4096 (lo16 @ [63:48])
  g1[2] = (int)((4096u >> 16) | (1024u << 16));            // tensor_dim0 hi | tensor_dim1 lo (1024)
  g1[3] = (int)(4096u << 16);                              // tensor_dim1 hi (0) | tile_dim0 = 4096
  g1[4] = (int)CH;                                         // tile_dim1 = CH, tile_dim2 = 0
  g1[5] = (int)(4096u * BAT);                              // tensor_dim0_stride = 524288
  g1[6] = 0;
  g1[7] = 0;
  i32x4 g2 = {0, 0, 0, 0};
  i32x4 g3 = {0, 0, 0, 0};
  i32x8 g4 = {0, 0, 0, 0, 0, 0, 0, 0};                     // extra group (clang-23 6-arg form), unused
  __builtin_amdgcn_tensor_load_to_lds(g0, g1, g2, g3, g4, 0);
}
#endif

__device__ __forceinline__ void chunk_issue(const float* __restrict__ scores, int b, int t0,
                                            float* dst, int tid) {
#if HAVE_TDM
  if (tid < 32) tdm_issue(scores, b, t0, (void*)dst);      // wave 0 issues the DMA
#else
  // synchronous fallback copy with manual row padding
  const float* gbase = scores + ((unsigned long long)t0 * BAT + (unsigned long long)b) * (TAGS * TAGS);
  for (int idx = tid; idx < CH * TAGS * TAGS; idx += 128) {
    const int s   = idx >> 12;
    const int rem = idx & 4095;
    const int i   = rem >> 6;
    const int j   = rem & 63;
    dst[s * TILE_DW + i * ROW_DW + j] =
        gbase[(unsigned long long)s * (BAT * TAGS * TAGS) + rem];
  }
#endif
}

__device__ __forceinline__ void chunk_wait(int tid) {
#if HAVE_TDM
  if (tid < 32) __builtin_amdgcn_s_wait_tensorcnt((short)0);
#else
  (void)tid;
#endif
}

__global__ __launch_bounds__(128) void crf_fwd_kernel(const float* __restrict__ scores,
                                                      const long long* __restrict__ target,
                                                      const unsigned char* __restrict__ mask,
                                                      float* __restrict__ per_batch) {
  __shared__ float tile[2][CHUNK_DW];   // double-buffered score tiles (padded rows)
  __shared__ float curbuf[2][TAGS];     // ping-pong log-partition vector
  __shared__ float part[TAGS];          // masked partition
  __shared__ float wbuf[TAGS];          // exp(cur - M) shared weights

  const int b    = blockIdx.x;
  const int tid  = threadIdx.x;
  const int lane = tid & 31;
  const int jc   = ((tid >> 5) << 4) + (lane & 15);   // this lane's output tag column

  float tg_acc = 0.f;

  chunk_issue(scores, b, 0, tile[0], tid);
  chunk_wait(tid);
  __syncthreads();

  for (int c = 0; c < NCHUNK; ++c) {
    if (c + 1 < NCHUNK) chunk_issue(scores, b, (c + 1) * CH, tile[(c + 1) & 1], tid);
    const float* cbuf = tile[c & 1];
    for (int s = 0; s < CH; ++s) {
      const int t = c * CH + s;
      const float* T = cbuf + s * TILE_DW;
      const unsigned char mk = mask[t * BAT + b];
      if (t == 0) {
        if (tid < TAGS) {
          const float v = T[START_TAG * ROW_DW + tid];
          curbuf[0][tid] = v;
          part[tid] = v;
        }
        if (tid == 0 && mk) {
          const long long fi = target[(long long)t * BAT + b];
          tg_acc += T[((int)(fi >> 6)) * ROW_DW + (int)(fi & 63)];
        }
        __syncthreads();
        continue;
      }
      const float* cur = curbuf[(t + 1) & 1];
      float* ncur      = curbuf[t & 1];

      // M = max_i cur[i] (wave32 shuffle reduce, replicated per wave)
      float m = fmaxf(cur[lane], cur[lane + 32]);
#pragma unroll
      for (int o = 16; o > 0; o >>= 1) m = fmaxf(m, __shfl_xor(m, o, 32));

      if (tid < TAGS) wbuf[tid] = __expf(cur[tid] - m);
      __syncthreads();

      // S[j] = sum_i w[i] * exp(scores[i][j])  via v_wmma_f32_16x16x32_f16, K split 2x32
      v8f acc = {0.f, 0.f, 0.f, 0.f, 0.f, 0.f, 0.f, 0.f};
#pragma unroll
      for (int chn = 0; chn < 2; ++chn) {
        // B 32x16 layout: VGPR g halves = K {2g,2g+1} (lanes 0-15) / {16+2g,16+2g+1} (lanes 16-31)
        const int ib = chn * 32 + ((lane < 16) ? 0 : 16);
        v16h B;
#pragma unroll
        for (int e = 0; e < 16; ++e)
          B[e] = (half_t)__expf(T[(ib + e) * ROW_DW + jc]);
        // A 16x32 layout: element e -> K = e (lanes 0-15) / 16+e (lanes 16-31), matching B
        const int ab = chn * 32 + ((lane < 16) ? 0 : 16);
        v16h A;
#pragma unroll
        for (int e = 0; e < 16; ++e) {
          const int g = e >> 1;
          const int K = ab + e;
          A[e] = (half_t)wbuf[K];
        }
        acc = __builtin_amdgcn_wmma_f32_16x16x32_f16(false, A, false, B, (short)0, acc,
                                                     false, false);
      }
      // D row 0 (element 0) holds S[jc] in every half-wave
      const float nc = m + __logf(acc[0]);
      if (lane < 16) {
        ncur[jc] = nc;
        if (mk) part[jc] = nc;
      }
      if (tid == 0 && mk) {
        const long long fi = target[(long long)t * BAT + b];
        tg_acc += T[((int)(fi >> 6)) * ROW_DW + (int)(fi & 63)];
      }
      __syncthreads();
    }
    chunk_wait(tid);
    __syncthreads();
  }
  if (tid == 0) per_batch[b] = part[END_TAG] - tg_acc;
}

__global__ __launch_bounds__(128) void crf_reduce_kernel(const float* __restrict__ per_batch,
                                                         float* __restrict__ out) {
  const int tid = threadIdx.x;
  float v = per_batch[tid];
#pragma unroll
  for (int o = 16; o > 0; o >>= 1) v += __shfl_xor(v, o, 32);
  __shared__ float ws[4];
  if ((tid & 31) == 0) ws[tid >> 5] = v;
  __syncthreads();
  if (tid == 0) out[0] = (ws[0] + ws[1] + ws[2] + ws[3]) * (1.0f / (float)BAT);
}

extern "C" void kernel_launch(void* const* d_in, const int* in_sizes, int n_in,
                              void* d_out, int out_size, void* d_ws, size_t ws_size,
                              hipStream_t stream) {
  (void)in_sizes; (void)n_in; (void)out_size; (void)ws_size;
  const float* scores         = (const float*)d_in[0];
  // d_in[1] corpus_mask unused by the reference loss
  const long long* target     = (const long long*)d_in[2];
  const unsigned char* maskb  = (const unsigned char*)d_in[3];
  float* per_batch            = (float*)d_ws;

  crf_fwd_kernel<<<BAT, 128, 0, stream>>>(scores, target, maskb, per_batch);
  crf_reduce_kernel<<<1, 128, 0, stream>>>(per_batch, (float*)d_out);
}
